// CrossDomainAspectLabelPropagation_17497696764163
// MI455X (gfx1250) — matrix operand order repaired and checked by reading, not compiled, not run on hardware
//
#include <hip/hip_runtime.h>
#include <stddef.h>

// ---------------------------------------------------------------------------
// CrossDomainAspectLabelPropagation — MI455X (gfx1250) bf16-WMMA pipeline. v3
//
// v3 change (driven by disasm): the v2 ping-pong rotation (a=an/b=bn) lowered
// to ~48 v_dual_mov_b32 per K-step because WMMA operands need contiguous even
// VGPR ranges. Now the K loop is unrolled by 64 with two fragment buffers that
// alternate in place (load buf1 / wmma buf0 / load buf0 / wmma buf1), K%64==0
// required (holds for 512/1024/2048). No copies, no hazard nops.
// ---------------------------------------------------------------------------

#define HID 1024
#define NDOM 8
#define NB 8192
#define LN_EPS 1e-5f

typedef unsigned short u16;
typedef unsigned int u32;

typedef __attribute__((ext_vector_type(16))) __bf16 v16bf;
typedef __attribute__((ext_vector_type(8)))  float  v8f;

// ---------------- small helpers ----------------

__device__ __forceinline__ u16 f2bf(float f) {
    u32 u = __float_as_uint(f);
    u32 r = u + 0x7FFFu + ((u >> 16) & 1u);   // round-to-nearest-even
    return (u16)(r >> 16);
}
__device__ __forceinline__ float bf2f(u16 h) {
    return __uint_as_float(((u32)h) << 16);
}

// A-matrix 16x32 bf16 fragment (per ISA: lanes0-15 K={0..7,16..23},
// lanes16-31 K={8..15,24..31}); rowp = &A[row*lda + k0]
__device__ __forceinline__ v16bf load_a_frag(const u16* rowp, int kh) {
    union { uint4 q[2]; v16bf v; } u;
    u.q[0] = *reinterpret_cast<const uint4*>(rowp + kh * 8);
    u.q[1] = *reinterpret_cast<const uint4*>(rowp + 16 + kh * 8);
    return u.v;
}
// B-matrix 32x16 bf16 fragment: same K order as A (lanes0-15 K={0..7,16..23},
// lanes16-31 K={8..15,24..31}); colp = &Bt[col*ldb + k0] with Bt = B^T stored [N][K]
__device__ __forceinline__ v16bf load_b_frag(const u16* colp, int kh) {
    union { uint4 q[2]; v16bf v; } u;
    const u16* p = colp + kh * 8;
    u.q[0] = *reinterpret_cast<const uint4*>(p);
    u.q[1] = *reinterpret_cast<const uint4*>(p + 16);
    return u.v;
}

struct Frags {
    v16bf a[2];
    v16bf b[4];
};

__device__ __forceinline__ void load_tiles(Frags& f, const u16* ap0, const u16* ap1,
                                           const u16* bp0, int ldb, int k0, int kh) {
    f.a[0] = load_a_frag(ap0 + k0, kh);
    f.a[1] = load_a_frag(ap1 + k0, kh);
    #pragma unroll
    for (int n = 0; n < 4; n++)
        f.b[n] = load_b_frag(bp0 + (size_t)(n * 16) * ldb + k0, kh);
}

__device__ __forceinline__ void wmma_tiles(v8f acc[2][4], const Frags& f) {
    #pragma unroll
    for (int m = 0; m < 2; m++)
        #pragma unroll
        for (int n = 0; n < 4; n++)
            acc[m][n] = __builtin_amdgcn_wmma_f32_16x16x32_bf16(
                false, f.a[m], false, f.b[n], (short)0, acc[m][n], false, false);
}

// ---------------- conversion kernels ----------------

__global__ __launch_bounds__(256) void k_f32_to_bf16(const float* __restrict__ x,
                                                     u16* __restrict__ y, long n) {
    long i = (long)blockIdx.x * blockDim.x + threadIdx.x;
    long stride = (long)gridDim.x * blockDim.x;
    for (; i < n; i += stride) y[i] = f2bf(x[i]);
}

// W [K,N] fp32 -> Wt [N,K] bf16 (batched over blockIdx.z, stride K*N)
__global__ __launch_bounds__(256) void k_transpose_bf16(const float* __restrict__ W,
                                                        u16* __restrict__ Wt,
                                                        int K, int N) {
    __shared__ float tile[32][33];
    size_t bstride = (size_t)K * N;
    const float* Wb = W + bstride * blockIdx.z;
    u16* Wtb = Wt + bstride * blockIdx.z;
    int n0 = blockIdx.x * 32, k0 = blockIdx.y * 32;
    int tx = threadIdx.x, ty = threadIdx.y;   // block (32,8)
    #pragma unroll
    for (int i = ty; i < 32; i += 8)
        tile[i][tx] = Wb[(size_t)(k0 + i) * N + n0 + tx];
    __syncthreads();
    #pragma unroll
    for (int i = ty; i < 32; i += 8)
        Wtb[(size_t)(n0 + i) * K + k0 + tx] = f2bf(tile[tx][i]);
}

// ---------------- dense bf16 WMMA GEMM ----------------
// C[M,N] = act(A[M,K] * B[K,N] + bias), A row-major bf16 (lda), Bt = B^T [N,K].
// Block: 128x128, 8 waves (4x2), wave tile 32x64 (2x4 WMMA 16x16 tiles).
// M, N multiples of 128; K multiple of 64. Templated epilogue.

template <int OUTF, int OUTB, int RELU>
__global__ __launch_bounds__(256) void k_gemm_bf16(
    const u16* __restrict__ A, int lda,
    const u16* __restrict__ Bt, int ldb,
    const float* __restrict__ bias,
    float* __restrict__ Cf, int ldcf,
    u16* __restrict__ Cb, int ldcb,
    int K) {
    const int lane = threadIdx.x & 31;
    const int wave = threadIdx.x >> 5;
    const int wm = wave & 3, wn = wave >> 2;
    const int rowBase = blockIdx.y * 128 + wm * 32;
    const int colBase = blockIdx.x * 128 + wn * 64;
    const int r16 = lane & 15, kh = lane >> 4;

    const u16* ap0 = A + (size_t)(rowBase + r16) * lda;
    const u16* ap1 = ap0 + (size_t)16 * lda;
    const u16* bp0 = Bt + (size_t)(colBase + r16) * ldb;

    v8f acc[2][4] = {};
    Frags f0, f1;
    load_tiles(f0, ap0, ap1, bp0, ldb, 0, kh);

    // main loop: two in-place buffers, no inter-buffer copies (K % 64 == 0)
    for (int k0 = 0; k0 < K - 64; k0 += 64) {
        __builtin_prefetch(ap0 + k0 + 128, 0, 3);
        __builtin_prefetch(bp0 + k0 + 128, 0, 3);
        load_tiles(f1, ap0, ap1, bp0, ldb, k0 + 32, kh);
        wmma_tiles(acc, f0);
        load_tiles(f0, ap0, ap1, bp0, ldb, k0 + 64, kh);
        wmma_tiles(acc, f1);
    }
    // tail (last 64 K): f0 holds K-64
    load_tiles(f1, ap0, ap1, bp0, ldb, K - 32, kh);
    wmma_tiles(acc, f0);
    wmma_tiles(acc, f1);

    // branch-free epilogue
    float bv[4];
    #pragma unroll
    for (int n = 0; n < 4; n++) bv[n] = bias[colBase + n * 16 + r16];

    #pragma unroll
    for (int m = 0; m < 2; m++)
        #pragma unroll
        for (int r = 0; r < 8; r++) {
            const int row = rowBase + m * 16 + r + kh * 8;
            float* cf = OUTF ? (Cf + (size_t)row * ldcf + colBase + r16) : nullptr;
            u16* cb = OUTB ? (Cb + (size_t)row * ldcb + colBase + r16) : nullptr;
            #pragma unroll
            for (int n = 0; n < 4; n++) {
                float v = acc[m][n][r] + bv[n];
                if (RELU) v = v > 0.f ? v : 0.f;
                if (OUTF) cf[n * 16] = v;
                if (OUTB) cb[n * 16] = f2bf(v);
            }
        }
}

// ---------------- routed (per-domain) bf16 WMMA GEMM ----------------
// blockIdx.z = expert id. Rows gathered via perm[offs[e]..offs[e+1]).
// Results scattered back to original row indices. K % 64 == 0.

template <int OUTF, int OUTB, int RELU>
__global__ __launch_bounds__(256) void k_gemm_routed_bf16(
    const u16* __restrict__ A, int lda,
    const u16* __restrict__ BtBase, int ldb, size_t wStride,
    const float* __restrict__ biasBase, int biasStride,
    float* __restrict__ Cf, int ldcf,
    u16* __restrict__ Cb, int ldcb,
    int K,
    const int* __restrict__ perm, const int* __restrict__ offs) {
    const int e = blockIdx.z;
    const int base = offs[e];
    const int cnt = offs[e + 1] - base;
    if ((int)(blockIdx.y * 128) >= cnt) return;

    const u16* Bt = BtBase + wStride * e;
    const float* bias = biasBase + (size_t)biasStride * e;

    const int lane = threadIdx.x & 31;
    const int wave = threadIdx.x >> 5;
    const int wm = wave & 3, wn = wave >> 2;
    const int lrBase = blockIdx.y * 128 + wm * 32;
    const int colBase = blockIdx.x * 128 + wn * 64;
    const int r16 = lane & 15, kh = lane >> 4;

    // gather source rows (clamped; out-of-range lanes load a duplicate row)
    const u16* ap0;
    const u16* ap1;
    {
        int lr0 = lrBase + r16;        lr0 = lr0 < cnt ? lr0 : cnt - 1;
        int lr1 = lrBase + 16 + r16;   lr1 = lr1 < cnt ? lr1 : cnt - 1;
        ap0 = A + (size_t)perm[base + lr0] * lda;
        ap1 = A + (size_t)perm[base + lr1] * lda;
    }
    const u16* bp0 = Bt + (size_t)(colBase + r16) * ldb;

    v8f acc[2][4] = {};
    Frags f0, f1;
    load_tiles(f0, ap0, ap1, bp0, ldb, 0, kh);

    for (int k0 = 0; k0 < K - 64; k0 += 64) {
        __builtin_prefetch(ap0 + k0 + 128, 0, 3);
        __builtin_prefetch(bp0 + k0 + 128, 0, 3);
        load_tiles(f1, ap0, ap1, bp0, ldb, k0 + 32, kh);
        wmma_tiles(acc, f0);
        load_tiles(f0, ap0, ap1, bp0, ldb, k0 + 64, kh);
        wmma_tiles(acc, f1);
    }
    load_tiles(f1, ap0, ap1, bp0, ldb, K - 32, kh);
    wmma_tiles(acc, f0);
    wmma_tiles(acc, f1);

    float bv[4];
    #pragma unroll
    for (int n = 0; n < 4; n++) bv[n] = bias[colBase + n * 16 + r16];

    #pragma unroll
    for (int m = 0; m < 2; m++)
        #pragma unroll
        for (int r = 0; r < 8; r++) {
            const int lr = lrBase + m * 16 + r + kh * 8;
            if (lr >= cnt) continue;
            const int row = perm[base + lr];
            float* cf = OUTF ? (Cf + (size_t)row * ldcf + colBase + r16) : nullptr;
            u16* cb = OUTB ? (Cb + (size_t)row * ldcb + colBase + r16) : nullptr;
            #pragma unroll
            for (int n = 0; n < 4; n++) {
                float v = acc[m][n][r] + bv[n];
                if (RELU) v = v > 0.f ? v : 0.f;
                if (OUTF) cf[n * 16] = v;
                if (OUTB) cb[n * 16] = f2bf(v);
            }
        }
}

// ---------------- LayerNorm (fp32 in; fp32 and/or bf16 out) ----------------
// One block (256 threads) per row of X [*, H]. Optional per-row expert params.

__global__ __launch_bounds__(256) void k_layernorm(
    const float* __restrict__ X,
    const float* __restrict__ gB, const float* __restrict__ beB,
    const int* __restrict__ dids,
    float* __restrict__ Fout,
    u16* __restrict__ Bout, int bStride) {
    __shared__ float rs[256], rq[256];
    const int row = blockIdx.x;
    const float* x = X + (size_t)row * HID;
    float s = 0.f, q = 0.f;
    for (int c = threadIdx.x; c < HID; c += 256) {
        float v = x[c];
        s += v; q += v * v;
    }
    rs[threadIdx.x] = s; rq[threadIdx.x] = q;
    __syncthreads();
    for (int st = 128; st > 0; st >>= 1) {
        if ((int)threadIdx.x < st) {
            rs[threadIdx.x] += rs[threadIdx.x + st];
            rq[threadIdx.x] += rq[threadIdx.x + st];
        }
        __syncthreads();
    }
    float mean = rs[0] * (1.f / HID);
    float var = rq[0] * (1.f / HID) - mean * mean;
    float rstd = rsqrtf(var + LN_EPS);
    int e = dids ? dids[row] : 0;
    const float* g = gB + (size_t)e * HID;
    const float* be = beB + (size_t)e * HID;
    for (int c = threadIdx.x; c < HID; c += 256) {
        float v = (x[c] - mean) * rstd * g[c] + be[c];
        if (Fout) Fout[(size_t)row * HID + c] = v;
        if (Bout) Bout[(size_t)row * bStride + c] = f2bf(v);
    }
}

// ---------------- routing: histogram / scan / scatter ----------------
// ints layout: [0..7]=counts, [8..16]=offsets(9), [17..24]=cursor

__global__ void k_hist(const int* __restrict__ ids, int* __restrict__ ints) {
    int t = blockIdx.x * blockDim.x + threadIdx.x;
    if (t < NB) atomicAdd(&ints[ids[t] & (NDOM - 1)], 1);
}
__global__ void k_scan(int* __restrict__ ints) {
    int* counts = ints;
    int* offs = ints + 8;
    int* cur = ints + 17;
    offs[0] = 0;
    for (int e = 0; e < NDOM; e++) {
        offs[e + 1] = offs[e] + counts[e];
        cur[e] = offs[e];
    }
}
__global__ void k_scatter(const int* __restrict__ ids, int* __restrict__ ints,
                          int* __restrict__ perm) {
    int t = blockIdx.x * blockDim.x + threadIdx.x;
    if (t < NB) {
        int e = ids[t] & (NDOM - 1);
        int p = atomicAdd(&ints[17 + e], 1);
        perm[p] = t;
    }
}

// ---------------- tiny-N head (VALU) ----------------
// Out[b,n] = A[b,:](bf16) . W[:,n](fp32) + bias[n]

__global__ __launch_bounds__(256) void k_small_gemm(
    const u16* __restrict__ A, int lda,
    const float* __restrict__ W, const float* __restrict__ bias,
    float* __restrict__ Out, int ldo, int M, int N, int K) {
    int t = blockIdx.x * blockDim.x + threadIdx.x;
    if (t >= M * N) return;
    int b = t / N, n = t - b * N;
    float s = bias[n];
    const u16* a = A + (size_t)b * lda;
    #pragma unroll 8
    for (int k = 0; k < K; k++) s += bf2f(a[k]) * W[(size_t)k * N + n];
    Out[(size_t)b * ldo + n] = s;
}

// ---------------- host-side launch ----------------

extern "C" void kernel_launch(void* const* d_in, const int* in_sizes, int n_in,
                              void* d_out, int out_size, void* d_ws, size_t ws_size,
                              hipStream_t stream) {
    (void)n_in; (void)out_size; (void)ws_size;

    const float* feats = (const float*)d_in[0];
    const int* dom_ids = (const int*)d_in[1];

    // Param pointer mapping: handle both insertion-order and sorted-key
    // pytree flattening; distinguished by in_sizes[2]
    // (sorted -> cls.W1 = 2*H*H ; insertion -> inv.W1 = H*H).
    const float *clsW1,*clsW2,*clsb1,*clsb2;
    const float *discW1,*discW2,*discb1,*discb2;
    const float *domW1,*domW2,*domb1,*domb2,*domg,*dombe;
    const float *invW1,*invW2,*invb1,*invb2,*invg,*invbe;
    const float *varW1,*varW2,*varb1,*varb2,*varg,*varbe;
    #define FP(i) ((const float*)d_in[(i)])
    if (in_sizes[2] == 2 * HID * HID) {
        // sorted keys: cls, disc, dom, inv, var ; within: W1,W2,b1,b2[,be,g]
        clsW1=FP(2);  clsW2=FP(3);  clsb1=FP(4);  clsb2=FP(5);
        discW1=FP(6); discW2=FP(7); discb1=FP(8); discb2=FP(9);
        domW1=FP(10); domW2=FP(11); domb1=FP(12); domb2=FP(13); dombe=FP(14); domg=FP(15);
        invW1=FP(16); invW2=FP(17); invb1=FP(18); invb2=FP(19); invbe=FP(20); invg=FP(21);
        varW1=FP(22); varW2=FP(23); varb1=FP(24); varb2=FP(25); varbe=FP(26); varg=FP(27);
    } else {
        // insertion order: inv, var, dom, disc, cls ; within: W1,b1,W2,b2[,g,be]
        invW1=FP(2);  invb1=FP(3);  invW2=FP(4);  invb2=FP(5);  invg=FP(6);  invbe=FP(7);
        varW1=FP(8);  varb1=FP(9);  varW2=FP(10); varb2=FP(11); varg=FP(12); varbe=FP(13);
        domW1=FP(14); domb1=FP(15); domW2=FP(16); domb2=FP(17); domg=FP(18); dombe=FP(19);
        discW1=FP(20); discb1=FP(21); discW2=FP(22); discb2=FP(23);
        clsW1=FP(24); clsb1=FP(25); clsW2=FP(26); clsb2=FP(27);
    }
    #undef FP

    // ---- workspace carve (256B aligned) ----
    size_t off = 0;
    auto carve = [&](size_t bytes) -> void* {
        void* p = (void*)((char*)d_ws + off);
        off += (bytes + 255) & ~(size_t)255;
        return p;
    };
    u16* xbf     = (u16*)carve((size_t)NB * HID * 2);       // features bf16
    u16* hbf     = (u16*)carve((size_t)NB * HID * 2);       // hidden bf16 (reused)
    u16* varbf   = (u16*)carve((size_t)NB * HID * 2);       // var encoder out bf16
    u16* combbf  = (u16*)carve((size_t)NB * 2 * HID * 2);   // [inv | dom] bf16
    u16* dischbf = (u16*)carve((size_t)NB * 512 * 2);       // disc hidden bf16
    u16* wInv1T  = (u16*)carve((size_t)HID * HID * 2);
    u16* wInv2T  = (u16*)carve((size_t)HID * HID * 2);
    u16* wVar1T  = (u16*)carve((size_t)HID * HID * 2);
    u16* wVar2T  = (u16*)carve((size_t)HID * HID * 2);
    u16* wDom1T  = (u16*)carve((size_t)NDOM * HID * HID * 2);
    u16* wDom2T  = (u16*)carve((size_t)NDOM * HID * HID * 2);
    u16* wDisc1T = (u16*)carve((size_t)512 * HID * 2);
    u16* wCls1T  = (u16*)carve((size_t)HID * 2 * HID * 2);
    float* tmpf  = (float*)carve((size_t)NB * HID * 4);     // pre-LN fp32
    int* perm    = (int*)carve((size_t)NB * 4);
    int* ints    = (int*)carve(128);

    float* out = (float*)d_out;
    float* outAspect = out;                                   // [B,3]
    float* outDpred  = out + (size_t)3 * NB;                  // [B,8]
    float* outInv    = out + (size_t)11 * NB;                 // [B,H]
    float* outDom    = out + (size_t)11 * NB + (size_t)NB * HID; // [B,H]

    dim3 blk256(256);
    dim3 blkT(32, 8);

    // 0) convert features
    k_f32_to_bf16<<<4096, blk256, 0, stream>>>(feats, xbf, (long)NB * HID);

    // 1) transpose+convert weights (W[K,N] -> Wt[N,K] bf16)
    k_transpose_bf16<<<dim3(32, 32, 1), blkT, 0, stream>>>(invW1, wInv1T, HID, HID);
    k_transpose_bf16<<<dim3(32, 32, 1), blkT, 0, stream>>>(invW2, wInv2T, HID, HID);
    k_transpose_bf16<<<dim3(32, 32, 1), blkT, 0, stream>>>(varW1, wVar1T, HID, HID);
    k_transpose_bf16<<<dim3(32, 32, 1), blkT, 0, stream>>>(varW2, wVar2T, HID, HID);
    k_transpose_bf16<<<dim3(32, 32, NDOM), blkT, 0, stream>>>(domW1, wDom1T, HID, HID);
    k_transpose_bf16<<<dim3(32, 32, NDOM), blkT, 0, stream>>>(domW2, wDom2T, HID, HID);
    k_transpose_bf16<<<dim3(16, 32, 1), blkT, 0, stream>>>(discW1, wDisc1T, HID, 512);
    k_transpose_bf16<<<dim3(32, 64, 1), blkT, 0, stream>>>(clsW1, wCls1T, 2 * HID, HID);

    const dim3 gFull(HID / 128, NB / 128);        // (8, 64)

    // 2) inv encoder
    k_gemm_bf16<0,1,1><<<gFull, blk256, 0, stream>>>(xbf, HID, wInv1T, HID, invb1,
                                                     nullptr, 0, hbf, HID, HID);
    k_gemm_bf16<1,0,0><<<gFull, blk256, 0, stream>>>(hbf, HID, wInv2T, HID, invb2,
                                                     tmpf, HID, nullptr, 0, HID);
    k_layernorm<<<NB, blk256, 0, stream>>>(tmpf, invg, invbe, nullptr,
                                           outInv, combbf, 2 * HID);

    // 3) var encoder
    k_gemm_bf16<0,1,1><<<gFull, blk256, 0, stream>>>(xbf, HID, wVar1T, HID, varb1,
                                                     nullptr, 0, hbf, HID, HID);
    k_gemm_bf16<1,0,0><<<gFull, blk256, 0, stream>>>(hbf, HID, wVar2T, HID, varb2,
                                                     tmpf, HID, nullptr, 0, HID);
    k_layernorm<<<NB, blk256, 0, stream>>>(tmpf, varg, varbe, nullptr,
                                           nullptr, varbf, HID);

    // 4) routing permutation
    hipMemsetAsync(ints, 0, 128, stream);
    k_hist<<<NB / 256, blk256, 0, stream>>>(dom_ids, ints);
    k_scan<<<1, 1, 0, stream>>>(ints);
    k_scatter<<<NB / 256, blk256, 0, stream>>>(dom_ids, ints, perm);

    // 5) routed expert encoder
    const dim3 gRoute(HID / 128, NB / 128, NDOM); // worst case one domain has all rows
    k_gemm_routed_bf16<0,1,1><<<gRoute, blk256, 0, stream>>>(
        varbf, HID, wDom1T, HID, (size_t)HID * HID, domb1, HID,
        nullptr, 0, hbf, HID, HID, perm, ints + 8);
    k_gemm_routed_bf16<1,0,0><<<gRoute, blk256, 0, stream>>>(
        hbf, HID, wDom2T, HID, (size_t)HID * HID, domb2, HID,
        tmpf, HID, nullptr, 0, HID, perm, ints + 8);
    k_layernorm<<<NB, blk256, 0, stream>>>(tmpf, domg, dombe, dom_ids,
                                           outDom, combbf + HID, 2 * HID);

    // 6) classifier: relu(comb @ clsW1 + b1) @ clsW2 + b2
    k_gemm_bf16<0,1,1><<<gFull, blk256, 0, stream>>>(combbf, 2 * HID, wCls1T, 2 * HID,
                                                     clsb1, nullptr, 0, hbf, HID, 2 * HID);
    k_small_gemm<<<(NB * 3 + 255) / 256, blk256, 0, stream>>>(
        hbf, HID, clsW2, clsb2, outAspect, 3, NB, 3, HID);

    // 7) discriminator: relu(inv @ discW1 + b1) @ discW2 + b2
    k_gemm_bf16<0,1,1><<<dim3(512 / 128, NB / 128), blk256, 0, stream>>>(
        combbf, 2 * HID, wDisc1T, HID, discb1,
        nullptr, 0, dischbf, 512, HID);
    k_small_gemm<<<(NB * NDOM + 255) / 256, blk256, 0, stream>>>(
        dischbf, 512, discW2, discb2, outDpred, NDOM, NB, NDOM, 512);
}
